// GPT_81827716923583
// MI455X (gfx1250) — matrix operand order repaired and checked by reading, not compiled, not run on hardware
//
#include <hip/hip_runtime.h>
#include <math.h>

// ---------------- problem constants (match reference) ----------------
#define E_DIM   512
#define S_LEN   256
#define B_SZ    16
#define NHEAD   8
#define HDIM    64
#define NROWS   (B_SZ * S_LEN)      // 4096
#define VSIZE   32000
#define NBLK    6
#define FF_DIM  2048
#define SCALE_ATT 22.627416997969522f   // sqrt(512), faithful to reference bug

typedef __attribute__((ext_vector_type(16))) __bf16 v16bf;
typedef __attribute__((ext_vector_type(8)))  float  v8f;

// float -> bf16 bits, round-to-nearest-even
__device__ __forceinline__ unsigned f2bf(float x) {
  union { float f; unsigned u; } v; v.f = x;
  unsigned r = v.u + 0x7FFFu + ((v.u >> 16) & 1u);
  return r >> 16;
}

// A-matrix (16x32 bf16) K index: lanes 0-15 hold K=0..15, lanes 16-31 K=16..31
__device__ __forceinline__ int kmapA(int lane, int j) {
  return ((lane & 16) ? 16 : 0) + j;
}
// B-matrix (32x16 bf16) K index: lanes 0-15 hold K=0..15, lanes 16-31 K=16..31
__device__ __forceinline__ int kmapB(int lane, int j) {
  return ((lane & 16) ? 16 : 0) + j;
}

__device__ __forceinline__ v8f wmma_bf16(v16bf a, v16bf b, v8f c) {
  return __builtin_amdgcn_wmma_f32_16x16x32_bf16(false, a, false, b,
                                                 (short)0, c, false, false);
}

// ---------------- embedding: x = tok[ids] + pos ----------------
__global__ __launch_bounds__(256)
void gpt_embed_kernel(const int* __restrict__ ids, const float* __restrict__ tok,
                      const float* __restrict__ pos, float* __restrict__ x) {
  int row = blockIdx.x;                 // 0..NROWS-1
  int s   = row & (S_LEN - 1);
  int id  = ids[row];
  for (int e = threadIdx.x; e < E_DIM; e += 256)
    x[(size_t)row * E_DIM + e] = tok[(size_t)id * E_DIM + e] + pos[(size_t)s * E_DIM + e];
}

// ---------------- layernorm (one block per row) ----------------
__global__ __launch_bounds__(256)
void gpt_ln_kernel(const float* __restrict__ x, const float* __restrict__ g,
                   const float* __restrict__ b, float* __restrict__ out) {
  __shared__ float red[256];
  int row = blockIdx.x;
  const float* xr = x + (size_t)row * E_DIM;
  float s = 0.f;
  for (int e = threadIdx.x; e < E_DIM; e += 256) s += xr[e];
  red[threadIdx.x] = s; __syncthreads();
  for (int off = 128; off > 0; off >>= 1) {
    if (threadIdx.x < off) red[threadIdx.x] += red[threadIdx.x + off];
    __syncthreads();
  }
  float mean = red[0] * (1.0f / E_DIM);
  __syncthreads();
  float vs = 0.f;
  for (int e = threadIdx.x; e < E_DIM; e += 256) { float d = xr[e] - mean; vs += d * d; }
  red[threadIdx.x] = vs; __syncthreads();
  for (int off = 128; off > 0; off >>= 1) {
    if (threadIdx.x < off) red[threadIdx.x] += red[threadIdx.x + off];
    __syncthreads();
  }
  float rstd = rsqrtf(red[0] * (1.0f / E_DIM) + 1e-5f);
  for (int e = threadIdx.x; e < E_DIM; e += 256)
    out[(size_t)row * E_DIM + e] = (xr[e] - mean) * rstd * g[e] + b[e];
}

// ---------------- WMMA GEMM: C[M,N] = A[M,K] @ B[K,N] + bias (+res)(relu) ----
// f32 in global, bf16 in LDS (fragment-order swizzled), f32 accumulate.
// Block tile 64x128, 8 waves (2x4) of 32x32 each, K-step 64, double-buffered LDS.
// Requires M%64==0, N%128==0, K%64==0 (true for all uses here).
template<bool RELU, bool RES>
__global__ __launch_bounds__(256)
void gpt_gemm_kernel(const float* __restrict__ A, const float* __restrict__ Bw,
                     const float* __restrict__ bias, const float* res,
                     float* C, int M, int N, int K) {
  // [stage][k-chunk of 32][16-row/col subtile][lane][frag element]
  __shared__ __attribute__((aligned(32))) __bf16 As[2][2][4][32][16];   // 16 KB
  __shared__ __attribute__((aligned(32))) __bf16 Bs[2][2][8][32][16];   // 32 KB

  const int tid  = threadIdx.x;
  const int lane = tid & 31;
  const int wave = tid >> 5;
  const int wm = wave >> 2, wn = wave & 3;        // 2 x 4 waves
  const int bM = blockIdx.y * 64, bN = blockIdx.x * 128;
  const int l15 = lane & 15, lhi = (lane >> 4) & 1;

  v8f acc[2][2];
  #pragma unroll
  for (int mi = 0; mi < 2; ++mi)
    #pragma unroll
    for (int ni = 0; ni < 2; ++ni)
      acc[mi][ni] = (v8f){0.f,0.f,0.f,0.f,0.f,0.f,0.f,0.f};

  // ---- stage one 64-deep K slice into LDS buffer `st` ----
  auto stage = [&](int st, int k0) {
    // A tile 64x64: 512 groups of 8 consecutive k; thread does 2 groups
    #pragma unroll
    for (int g2 = 0; g2 < 2; ++g2) {
      int g   = tid + g2 * 256;            // 0..511
      int row = g >> 3;                    // 0..63
      int c0  = (g & 7) * 8;               // 0..56
      int kc  = c0 >> 5;                   // which 32-chunk
      int kl  = c0 & 31;
      int aln = (row & 15) + ((kl & 16) ? 16 : 0);
      int j0  = (kl & 8) ? 8 : 0;
      const float4* ap = (const float4*)(A + (size_t)(bM + row) * K + k0 + c0);
      float4 fa = ap[0], fb = ap[1];
      uint4 pk;
      pk.x = f2bf(fa.x) | (f2bf(fa.y) << 16);
      pk.y = f2bf(fa.z) | (f2bf(fa.w) << 16);
      pk.z = f2bf(fb.x) | (f2bf(fb.y) << 16);
      pk.w = f2bf(fb.z) | (f2bf(fb.w) << 16);
      *(uint4*)&As[st][kc][row >> 4][aln][j0] = pk;
      if (k0 + 64 < K)
        __builtin_prefetch(A + (size_t)(bM + row) * K + k0 + 64 + c0, 0, 1);
    }
    // B tile 64x128: 512 (column, 16-k-group) pairs; thread does 2
    #pragma unroll
    for (int g2 = 0; g2 < 2; ++g2) {
      int g   = tid + g2 * 256;            // 0..511
      int n   = g & 127;
      int kg  = g >> 7;                    // 0..3 (16-k groups)
      int kc  = kg >> 1;
      int bln = (n & 15) + ((kg & 1) ? 16 : 0);
      const float* bp = Bw + (size_t)(k0 + kg * 16) * N + bN + n;
      unsigned pkB[8];
      #pragma unroll
      for (int q = 0; q < 16; q += 2) {
        unsigned lo = f2bf(bp[(size_t)q * N]);
        unsigned hi = f2bf(bp[(size_t)(q + 1) * N]);
        pkB[q >> 1] = lo | (hi << 16);
      }
      uint4* dst = (uint4*)&Bs[st][kc][n >> 4][bln][0];
      dst[0] = make_uint4(pkB[0], pkB[1], pkB[2], pkB[3]);
      dst[1] = make_uint4(pkB[4], pkB[5], pkB[6], pkB[7]);
      if (k0 + 64 < K)
        __builtin_prefetch(Bw + (size_t)(k0 + 64 + kg * 16) * N + bN + n, 0, 1);
    }
  };

  // ---- 8 WMMAs against LDS buffer `st` ----
  auto compute = [&](int st) {
    #pragma unroll
    for (int kc = 0; kc < 2; ++kc) {
      v16bf af[2], bfr[2];
      #pragma unroll
      for (int mi = 0; mi < 2; ++mi)
        af[mi] = *(const v16bf*)&As[st][kc][wm * 2 + mi][lane][0];
      #pragma unroll
      for (int ni = 0; ni < 2; ++ni)
        bfr[ni] = *(const v16bf*)&Bs[st][kc][wn * 2 + ni][lane][0];
      #pragma unroll
      for (int mi = 0; mi < 2; ++mi)
        #pragma unroll
        for (int ni = 0; ni < 2; ++ni)
          acc[mi][ni] = wmma_bf16(af[mi], bfr[ni], acc[mi][ni]);
    }
  };

  stage(0, 0);
  __syncthreads();
  int buf = 0;
  for (int k0 = 0; k0 < K; k0 += 64) {
    if (k0 + 64 < K) stage(buf ^ 1, k0 + 64);   // overlap next-tile loads with WMMAs
    compute(buf);
    __syncthreads();
    buf ^= 1;
  }

  // ---- epilogue: bias (+relu) (+residual) ----
  #pragma unroll
  for (int mi = 0; mi < 2; ++mi)
    #pragma unroll
    for (int ni = 0; ni < 2; ++ni) {
      #pragma unroll
      for (int i = 0; i < 8; ++i) {
        int row = bM + wm * 32 + mi * 16 + i + lhi * 8;
        int col = bN + wn * 32 + ni * 16 + l15;
        float v = acc[mi][ni][i] + bias[col];
        if (RELU) v = fmaxf(v, 0.f);
        if (RES)  v += res[(size_t)row * N + col];
        C[(size_t)row * N + col] = v;
      }
    }
}

// ---------------- flash-attention (one wave per (b*h, 16-query tile)) --------
__global__ __launch_bounds__(32)
void gpt_attn_kernel(const float* __restrict__ q, const float* __restrict__ k,
                     const float* __restrict__ v, float* __restrict__ ctx) {
  __shared__ __attribute__((aligned(32))) __bf16 sP[16][32];
  const int lane = threadIdx.x;
  const int qt = blockIdx.x;              // query tile 0..15
  const int bh = blockIdx.y;              // 0..B*H-1
  const int b = bh >> 3, h = bh & 7;
  const int l15 = lane & 15, lhi = (lane >> 4) & 1;

  const float* qb = q   + ((size_t)b * S_LEN) * E_DIM + h * HDIM;
  const float* kb = k   + ((size_t)b * S_LEN) * E_DIM + h * HDIM;
  const float* vb = v   + ((size_t)b * S_LEN) * E_DIM + h * HDIM;
  float*       cb = ctx + ((size_t)b * S_LEN) * E_DIM + h * HDIM;

  const v8f z8 = (v8f){0.f,0.f,0.f,0.f,0.f,0.f,0.f,0.f};

  // Q fragments: A-layout, two d-chunks of 32 over HD=64
  v16bf aq[2];
  #pragma unroll
  for (int c = 0; c < 2; ++c) {
    #pragma unroll
    for (int j = 0; j < 16; ++j) {
      int d = c * 32 + kmapA(lane, j);
      aq[c][j] = (__bf16)qb[(size_t)(qt * 16 + l15) * E_DIM + d];
    }
  }

  float m[8], lsum[8];
  v8f O[4];
  #pragma unroll
  for (int i = 0; i < 8; ++i) { m[i] = -3.0e38f; lsum[i] = 0.f; }
  #pragma unroll
  for (int t = 0; t < 4; ++t) O[t] = z8;

  const int nCh = ((qt + 1) * 16 + 31) >> 5;    // 32-key chunks (causal)
  for (int ch = 0; ch < nCh; ++ch) {
    const int keyBase = ch * 32;

    // scores: 16q x 32keys. Four INDEPENDENT WMMAs (no D->C chain -> no v_nop
    // hazard stalls), then VALU adds to combine the two d-chunks.
    v16bf bk[2][2];
    #pragma unroll
    for (int sub = 0; sub < 2; ++sub) {
      int key = keyBase + sub * 16 + l15;
      #pragma unroll
      for (int c = 0; c < 2; ++c) {
        #pragma unroll
        for (int j = 0; j < 16; ++j) {
          int d = c * 32 + kmapB(lane, j);
          bk[sub][c][j] = (__bf16)kb[(size_t)key * E_DIM + d];
        }
      }
    }
    v8f s00 = wmma_bf16(aq[0], bk[0][0], z8);
    v8f s10 = wmma_bf16(aq[0], bk[1][0], z8);
    v8f s01 = wmma_bf16(aq[1], bk[0][1], z8);
    v8f s11 = wmma_bf16(aq[1], bk[1][1], z8);
    v8f st[2];
    st[0] = s00 + s01;
    st[1] = s10 + s11;

    // scale, causal mask, online softmax state update
    float mnew[8];
    #pragma unroll
    for (int i = 0; i < 8; ++i) {
      int qrow = qt * 16 + i + lhi * 8;
      float s0 = st[0][i] * SCALE_ATT;
      float s1 = st[1][i] * SCALE_ATT;
      if (keyBase      + l15 > qrow) s0 = -3.0e38f;
      if (keyBase + 16 + l15 > qrow) s1 = -3.0e38f;
      st[0][i] = s0; st[1][i] = s1;
      float vmx = fmaxf(s0, s1);
      #pragma unroll
      for (int off = 1; off < 16; off <<= 1)
        vmx = fmaxf(vmx, __shfl_xor(vmx, off, 32));
      mnew[i] = fmaxf(m[i], vmx);
    }
    #pragma unroll
    for (int i = 0; i < 8; ++i) {
      float f = __expf(m[i] - mnew[i]);
      lsum[i] *= f;
      #pragma unroll
      for (int t = 0; t < 4; ++t) O[t][i] *= f;
      float p0 = __expf(st[0][i] - mnew[i]);
      float p1 = __expf(st[1][i] - mnew[i]);
      float ps = p0 + p1;
      #pragma unroll
      for (int off = 1; off < 16; off <<= 1)
        ps += __shfl_xor(ps, off, 32);
      lsum[i] += ps;
      m[i] = mnew[i];
      sP[i + lhi * 8][l15]      = (__bf16)p0;   // D-layout -> LDS
      sP[i + lhi * 8][16 + l15] = (__bf16)p1;
    }
    __syncthreads();

    // P as A-fragment, V as B-fragment; O += P @ V (4 independent chains)
    v16bf pa;
    #pragma unroll
    for (int j = 0; j < 16; ++j)
      pa[j] = sP[l15][kmapA(lane, j)];
    #pragma unroll
    for (int t = 0; t < 4; ++t) {
      v16bf bv;
      #pragma unroll
      for (int j = 0; j < 16; ++j) {
        int key = keyBase + kmapB(lane, j);
        bv[j] = (__bf16)vb[(size_t)key * E_DIM + t * 16 + l15];
      }
      O[t] = wmma_bf16(pa, bv, O[t]);
    }
    __syncthreads();
  }

  // normalize and write ctx
  #pragma unroll
  for (int i = 0; i < 8; ++i) {
    float inv = 1.f / lsum[i];
    int s = qt * 16 + i + lhi * 8;
    #pragma unroll
    for (int t = 0; t < 4; ++t)
      cb[(size_t)s * E_DIM + t * 16 + l15] = O[t][i] * inv;
  }
}

// ---------------- per-row log-softmax NLL ----------------
__global__ __launch_bounds__(256)
void gpt_rowloss_kernel(const float* __restrict__ logits, const int* __restrict__ tgt,
                        float* __restrict__ rowloss) {
  __shared__ float red[256];
  int row = blockIdx.x;
  const float* lr = logits + (size_t)row * VSIZE;
  float mx = -3.0e38f;
  for (int i = threadIdx.x; i < VSIZE; i += 256) mx = fmaxf(mx, lr[i]);
  red[threadIdx.x] = mx; __syncthreads();
  for (int off = 128; off > 0; off >>= 1) {
    if (threadIdx.x < off) red[threadIdx.x] = fmaxf(red[threadIdx.x], red[threadIdx.x + off]);
    __syncthreads();
  }
  float gmx = red[0]; __syncthreads();
  float se = 0.f;
  for (int i = threadIdx.x; i < VSIZE; i += 256) se += __expf(lr[i] - gmx);
  red[threadIdx.x] = se; __syncthreads();
  for (int off = 128; off > 0; off >>= 1) {
    if (threadIdx.x < off) red[threadIdx.x] += red[threadIdx.x + off];
    __syncthreads();
  }
  if (threadIdx.x == 0) {
    int t = tgt[row];
    rowloss[row] = -(lr[t] - gmx - logf(red[0]));
  }
}

__global__ __launch_bounds__(256)
void gpt_lossreduce_kernel(const float* __restrict__ rowloss, float* __restrict__ out) {
  __shared__ float red[256];
  float s = 0.f;
  for (int i = threadIdx.x; i < NROWS; i += 256) s += rowloss[i];
  red[threadIdx.x] = s; __syncthreads();
  for (int off = 128; off > 0; off >>= 1) {
    if (threadIdx.x < off) red[threadIdx.x] += red[threadIdx.x + off];
    __syncthreads();
  }
  if (threadIdx.x == 0) out[0] = red[0] * (1.0f / NROWS);
}

// ---------------- host-side orchestration ----------------
extern "C" void kernel_launch(void* const* d_in, const int* in_sizes, int n_in,
                              void* d_out, int out_size, void* d_ws, size_t ws_size,
                              hipStream_t stream) {
  (void)in_sizes; (void)n_in; (void)out_size; (void)ws_size;
  const int*   ids = (const int*)d_in[0];
  const int*   tgt = (const int*)d_in[1];
  const float* tok = (const float*)d_in[2];
  const float* pos = (const float*)d_in[3];
  // per-block params: Wq,bq,Wk,bk,Wv,bv,Wo,bo,W1,b1,W2,b2,g1,be1,g2,be2
  auto P = [&](int l, int idx) { return (const float*)d_in[4 + l * 16 + idx]; };
  const float* gf   = (const float*)d_in[4 + NBLK * 16 + 0];
  const float* bfp  = (const float*)d_in[4 + NBLK * 16 + 1];
  const float* Wout = (const float*)d_in[4 + NBLK * 16 + 2];
  const float* bout = (const float*)d_in[4 + NBLK * 16 + 3];

  const size_t ACT = (size_t)NROWS * E_DIM;            // 2,097,152 floats
  float* ws  = (float*)d_ws;
  float* x   = ws;
  float* hbuf= x   + ACT;
  float* qb  = hbuf+ ACT;
  float* kb  = qb  + ACT;
  float* vbuf= kb  + ACT;
  float* ctx = vbuf+ ACT;
  float* ff1 = ctx + ACT;                               // NROWS x FF_DIM
  float* rls = ff1 + (size_t)NROWS * FF_DIM;            // NROWS

  const dim3 blk256(256), blk32(32);
  const dim3 gE(NROWS);                                 // per-row kernels
  const dim3 gQKV(E_DIM / 128, NROWS / 64);             // 4 x 64
  const dim3 gFF1(FF_DIM / 128, NROWS / 64);            // 16 x 64
  const dim3 gOut(VSIZE / 128, NROWS / 64);             // 250 x 64
  const dim3 gAttn(S_LEN / 16, B_SZ * NHEAD);           // 16 x 128

  gpt_embed_kernel<<<gE, blk256, 0, stream>>>(ids, tok, pos, x);

  for (int l = 0; l < NBLK; ++l) {
    gpt_ln_kernel<<<gE, blk256, 0, stream>>>(x, P(l, 12), P(l, 13), hbuf);
    gpt_gemm_kernel<false, false><<<gQKV, blk256, 0, stream>>>(hbuf, P(l, 0), P(l, 1), nullptr, qb,   NROWS, E_DIM, E_DIM);
    gpt_gemm_kernel<false, false><<<gQKV, blk256, 0, stream>>>(hbuf, P(l, 2), P(l, 3), nullptr, kb,   NROWS, E_DIM, E_DIM);
    gpt_gemm_kernel<false, false><<<gQKV, blk256, 0, stream>>>(hbuf, P(l, 4), P(l, 5), nullptr, vbuf, NROWS, E_DIM, E_DIM);
    gpt_attn_kernel<<<gAttn, blk32, 0, stream>>>(qb, kb, vbuf, ctx);
    gpt_gemm_kernel<false, true ><<<gQKV, blk256, 0, stream>>>(ctx, P(l, 6), P(l, 7), x, x, NROWS, E_DIM, E_DIM);
    gpt_ln_kernel<<<gE, blk256, 0, stream>>>(x, P(l, 14), P(l, 15), hbuf);
    gpt_gemm_kernel<true,  false><<<gFF1, blk256, 0, stream>>>(hbuf, P(l, 8), P(l, 9), nullptr, ff1, NROWS, FF_DIM, E_DIM);
    gpt_gemm_kernel<false, true ><<<gQKV, blk256, 0, stream>>>(ff1, P(l, 10), P(l, 11), x, x, NROWS, E_DIM, FF_DIM);
  }

  gpt_ln_kernel<<<gE, blk256, 0, stream>>>(x, gf, bfp, hbuf);
  float* logits = (float*)d_out;
  gpt_gemm_kernel<false, false><<<gOut, blk256, 0, stream>>>(hbuf, Wout, bout, nullptr, logits, NROWS, VSIZE, E_DIM);

  gpt_rowloss_kernel<<<gE, blk256, 0, stream>>>(logits, tgt, rls);
  gpt_lossreduce_kernel<<<dim3(1), blk256, 0, stream>>>(rls, logits + (size_t)NROWS * VSIZE);
}
